// Head_9294309228559
// MI455X (gfx1250) — matrix operand order repaired and checked by reading, not compiled, not run on hardware
//
#include <hip/hip_runtime.h>
#include <hip/hip_bf16.h>
#include <stdint.h>

#define BATCH 4
#define SEQ   2048
#define EMB   1024
#define HD    128

typedef __attribute__((ext_vector_type(16))) __bf16 v16bf;
typedef __attribute__((ext_vector_type(8)))  __bf16 v8bf;
typedef __attribute__((ext_vector_type(8)))  float  v8f;
typedef __attribute__((ext_vector_type(4)))  unsigned int v4u;
typedef __attribute__((ext_vector_type(8)))  int v8i;
typedef __attribute__((ext_vector_type(4)))  int v4i;

static __device__ __forceinline__ v16bf make_afrag(v8bf lo, v8bf hi) {
  return __builtin_shufflevector(lo, hi, 0,1,2,3,4,5,6,7,8,9,10,11,12,13,14,15);
}
static __device__ __forceinline__ v8f wmma_bf16(v16bf a, v16bf b, v8f c) {
  return __builtin_amdgcn_wmma_f32_16x16x32_bf16(false, a, false, b, (short)0, c, false, false);
}

// ---------------- Kernel 0a: x f32 -> bf16 ----------------
__global__ void cvt_x_kernel(const float* __restrict__ x, __bf16* __restrict__ xb, int n) {
  int i = blockIdx.x * blockDim.x + threadIdx.x;
  int stride = gridDim.x * blockDim.x;
  for (; i < n; i += stride) xb[i] = (__bf16)x[i];
}

// ---------------- Kernel 0b: W [E][H] f32 -> Wt [H][E] bf16 ----------------
__global__ void cvt_w_kernel(const float* __restrict__ Wq, const float* __restrict__ Wk,
                             const float* __restrict__ Wv,
                             __bf16* __restrict__ wtq, __bf16* __restrict__ wtk,
                             __bf16* __restrict__ wtv) {
  int i = blockIdx.x * blockDim.x + threadIdx.x;
  if (i >= EMB * HD) return;
  int e = i / HD, h = i % HD;
  int o = h * EMB + e;
  wtq[o] = (__bf16)Wq[i];
  wtk[o] = (__bf16)Wk[i];
  wtv[o] = (__bf16)Wv[i];
}

// ---------------- Kernel 1: q/k/v projection via WMMA, x-tile via TDM ----------------
// grid = (B*S)/16 workgroups, 256 threads (8 waves); wave w owns H-tile w.
__global__ __launch_bounds__(256) void proj_kernel(
    const __bf16* __restrict__ xb,
    const __bf16* __restrict__ wtq, const __bf16* __restrict__ wtk,
    const __bf16* __restrict__ wtv,
    __bf16* __restrict__ qb, __bf16* __restrict__ kb, __bf16* __restrict__ vT) {
  __shared__ __align__(128) __bf16 xs[16 * EMB];   // 32 KB tile
  const int wave = threadIdx.x >> 5;
  const int lane = threadIdx.x & 31;
  const int rowBase = blockIdx.x * 16;             // row in [0, B*S)

  if (wave == 0) {
    // Tensor Data Mover: 2D tile, 16 rows x 1024 bf16, row stride 1024 elems.
    uint64_t gaddr = (uint64_t)(uintptr_t)(xb + (size_t)rowBase * EMB);
    uint32_t laddr = (uint32_t)(uintptr_t)(&xs[0]);
    v4u g0;
    g0[0] = 1u;                                        // count=1 valid descriptor
    g0[1] = laddr;                                     // lds_addr
    g0[2] = (uint32_t)gaddr;                           // global_addr lo
    g0[3] = (uint32_t)((gaddr >> 32) & 0x01FFFFFFu) | (2u << 30); // addr hi | type=2
    v8i g1;
    g1[0] = (int)(1u << 16);                           // data_size = 2 bytes
    g1[1] = (int)((uint32_t)EMB << 16);                // tensor_dim0 lo16 (bits 63:48)
    g1[2] = (int)(16u << 16);                          // dim0 hi=0 | tensor_dim1=16
    g1[3] = (int)((uint32_t)EMB << 16);                // dim1 hi=0 | tile_dim0=1024
    g1[4] = 16;                                        // tile_dim1=16, tile_dim2=0
    g1[5] = EMB;                                       // tensor_dim0_stride lo32
    g1[6] = 0;
    g1[7] = 0;
    v4i gz4 = {0, 0, 0, 0};
    v8i gz8 = {0, 0, 0, 0, 0, 0, 0, 0};
    __builtin_amdgcn_tensor_load_to_lds(g0, g1, gz4, gz4, gz8, 0);
    __builtin_amdgcn_s_wait_tensorcnt(0);
  }
  __syncthreads();

  const int m = lane & 15;
  const int half = lane >> 4;
  const int hcol = wave * 16 + m;   // B-fragment column

  v8f accq = {0.f,0.f,0.f,0.f,0.f,0.f,0.f,0.f};
  v8f acck = accq, accv = accq;

  const __bf16* wq = wtq + (size_t)hcol * EMB;
  const __bf16* wk = wtk + (size_t)hcol * EMB;
  const __bf16* wv = wtv + (size_t)hcol * EMB;

  for (int k0 = 0; k0 < EMB; k0 += 32) {
    v8bf lo = *(const v8bf*)(&xs[m * EMB + k0 + half * 16]);
    v8bf hi = *(const v8bf*)(&xs[m * EMB + k0 + half * 16 + 8]);
    v16bf a = make_afrag(lo, hi);
    v16bf bq = *(const v16bf*)(wq + k0 + half * 16);
    v16bf bk = *(const v16bf*)(wk + k0 + half * 16);
    v16bf bv = *(const v16bf*)(wv + k0 + half * 16);
    accq = wmma_bf16(a, bq, accq);
    acck = wmma_bf16(a, bk, acck);
    accv = wmma_bf16(a, bv, accv);
  }

  const float scale = 0.08838834764831845f;  // 1/sqrt(128), folded into q
  #pragma unroll
  for (int r = 0; r < 8; ++r) {
    int row = rowBase + r + half * 8;
    int b = row / SEQ;
    int s = row - b * SEQ;
    qb[(size_t)row * HD + hcol] = (__bf16)(accq[r] * scale);
    kb[(size_t)row * HD + hcol] = (__bf16)acck[r];
    vT[((size_t)b * HD + hcol) * SEQ + s] = (__bf16)accv[r];  // V transposed
  }
}

// ---------------- Kernel 2: causal flash attention, 1 wave per 16-row q tile ----------------
__global__ __launch_bounds__(128) void attn_kernel(
    const __bf16* __restrict__ qb, const __bf16* __restrict__ kb,
    const __bf16* __restrict__ vT, float* __restrict__ out) {
  __shared__ __align__(128) __bf16 pstage[4][16 * 32];  // per-wave P staging (1 KB each)
  const int wave = threadIdx.x >> 5;
  const int lane = threadIdx.x & 31;
  const int tile = blockIdx.x * 4 + wave;   // 0..511
  const int b  = tile >> 7;                 // tile / (SEQ/16)
  const int q0 = (tile & 127) * 16;         // q row base within batch

  const __bf16* Q  = qb + ((size_t)b * SEQ + q0) * HD;
  const __bf16* Kp = kb + (size_t)b * SEQ * HD;
  const __bf16* Vp = vT + (size_t)b * HD * SEQ;   // [H][S]

  const int m = lane & 15;
  const int half = lane >> 4;
  const float NEG_INF = -__builtin_inff();

  // Q fragments for the 4 K-steps over H=128
  v16bf qf[4];
  #pragma unroll
  for (int i = 0; i < 4; ++i) {
    const __bf16* p = Q + m * HD + i * 32 + half * 16;
    v8bf lo = *(const v8bf*)p;
    v8bf hi = *(const v8bf*)(p + 8);
    qf[i] = make_afrag(lo, hi);
  }

  v8f O[8];
  #pragma unroll
  for (int t = 0; t < 8; ++t) O[t] = (v8f){0,0,0,0,0,0,0,0};
  float mrow[8], lrow[8];
  #pragma unroll
  for (int r = 0; r < 8; ++r) { mrow[r] = NEG_INF; lrow[r] = 0.f; }

  __bf16* ps = &pstage[wave][0];
  const int nb = (q0 + 47) >> 5;   // number of 32-key causal blocks

  for (int jb = 0; jb < nb; ++jb) {
    const int kbase = jb * 32;
    v8f S0 = (v8f){0,0,0,0,0,0,0,0};
    v8f S1 = (v8f){0,0,0,0,0,0,0,0};
    #pragma unroll
    for (int i = 0; i < 4; ++i) {
      int k0 = i * 32;
      v16bf b0 = *(const v16bf*)(Kp + (size_t)(kbase + m) * HD + k0 + half * 16);
      v16bf b1 = *(const v16bf*)(Kp + (size_t)(kbase + 16 + m) * HD + k0 + half * 16);
      S0 = wmma_bf16(qf[i], b0, S0);
      S1 = wmma_bf16(qf[i], b1, S1);
    }
    if (jb == nb - 1) {  // only the diagonal block needs causal masking
      #pragma unroll
      for (int r = 0; r < 8; ++r) {
        int qrow = q0 + r + half * 8;
        if (kbase + m > qrow)      S0[r] = NEG_INF;
        if (kbase + 16 + m > qrow) S1[r] = NEG_INF;
      }
    }
    float alpha[8];
    #pragma unroll
    for (int r = 0; r < 8; ++r) {
      float bm = fmaxf(S0[r], S1[r]);
      #pragma unroll
      for (int off = 1; off < 16; off <<= 1)
        bm = fmaxf(bm, __shfl_xor(bm, off, 32));
      float nm = fmaxf(mrow[r], bm);
      alpha[r] = __expf(mrow[r] - nm);
      mrow[r] = nm;
    }
    #pragma unroll
    for (int r = 0; r < 8; ++r) {
      float p0 = __expf(S0[r] - mrow[r]);
      float p1 = __expf(S1[r] - mrow[r]);
      float rs = p0 + p1;
      #pragma unroll
      for (int off = 1; off < 16; off <<= 1)
        rs += __shfl_xor(rs, off, 32);
      lrow[r] = lrow[r] * alpha[r] + rs;
      int prow = r + half * 8;
      ps[prow * 32 + m]      = (__bf16)p0;
      ps[prow * 32 + 16 + m] = (__bf16)p1;
    }
    #pragma unroll
    for (int t = 0; t < 8; ++t)
      #pragma unroll
      for (int r = 0; r < 8; ++r) O[t][r] *= alpha[r];

    // accumulator layout -> A-fragment layout via per-wave LDS (same-wave DS is in-order)
    v8bf lo = *(const v8bf*)(ps + m * 32 + half * 16);
    v8bf hi = *(const v8bf*)(ps + m * 32 + half * 16 + 8);
    v16bf pa = make_afrag(lo, hi);
    #pragma unroll
    for (int t = 0; t < 8; ++t) {
      v16bf bv = *(const v16bf*)(Vp + (size_t)(t * 16 + m) * SEQ + kbase + half * 16);
      O[t] = wmma_bf16(pa, bv, O[t]);
    }
  }

  #pragma unroll
  for (int r = 0; r < 8; ++r) lrow[r] = 1.0f / lrow[r];
  float* op = out + ((size_t)b * SEQ + q0) * HD;
  #pragma unroll
  for (int t = 0; t < 8; ++t)
    #pragma unroll
    for (int r = 0; r < 8; ++r)
      op[(size_t)(r + half * 8) * HD + t * 16 + m] = O[t][r] * lrow[r];
}

// ---------------- Launch ----------------
extern "C" void kernel_launch(void* const* d_in, const int* in_sizes, int n_in,
                              void* d_out, int out_size, void* d_ws, size_t ws_size,
                              hipStream_t stream) {
  const float* x  = (const float*)d_in[0];
  const float* Wq = (const float*)d_in[1];
  const float* Wk = (const float*)d_in[2];
  const float* Wv = (const float*)d_in[3];
  float* out = (float*)d_out;

  char* ws = (char*)d_ws;
  // bytes, 256-aligned offsets
  __bf16* xb  = (__bf16*)(ws);                         // 16,777,216 B
  __bf16* wtq = (__bf16*)(ws + 16777216);              //    262,144 B
  __bf16* wtk = (__bf16*)(ws + 16777216 + 262144);
  __bf16* wtv = (__bf16*)(ws + 16777216 + 524288);
  __bf16* qb  = (__bf16*)(ws + 17563648);              //  2,097,152 B
  __bf16* kb  = (__bf16*)(ws + 17563648 + 2097152);
  __bf16* vT  = (__bf16*)(ws + 17563648 + 4194304);

  const int nx = BATCH * SEQ * EMB;
  cvt_x_kernel<<<4096, 256, 0, stream>>>(x, xb, nx);
  cvt_w_kernel<<<(EMB * HD + 255) / 256, 256, 0, stream>>>(Wq, Wk, Wv, wtq, wtk, wtv);
  proj_kernel<<<(BATCH * SEQ) / 16, 256, 0, stream>>>(xb, wtq, wtk, wtv, qb, kb, vT);
  attn_kernel<<<(BATCH * SEQ) / 64, 128, 0, stream>>>(qb, kb, vT, out);
}
